// QuantDense_11476152614999
// MI455X (gfx1250) — 8-bit matrix operand order repaired and checked by reading, compiled, not run on hardware
//
#include <hip/hip_runtime.h>
#include <stdint.h>

typedef __attribute__((ext_vector_type(8))) int v8i;

#define EPS 1e-8f
#define QN  127.0f

__global__ __launch_bounds__(256) void maxabs_kernel(const float4* __restrict__ x,
                                                     long n4,
                                                     unsigned* __restrict__ slot) {
    float m = 0.0f;
    long stride = (long)gridDim.x * blockDim.x;
    for (long i = (long)blockIdx.x * blockDim.x + threadIdx.x; i < n4; i += stride) {
        float4 v = x[i];
        m = fmaxf(m, fmaxf(fmaxf(fabsf(v.x), fabsf(v.y)),
                           fmaxf(fabsf(v.z), fabsf(v.w))));
    }
    __shared__ float red[256];
    int t = threadIdx.x;
    red[t] = m;
    __syncthreads();
    for (int s = 128; s > 0; s >>= 1) {
        if (t < s) red[t] = fmaxf(red[t], red[t + s]);
        __syncthreads();
    }
    if (t == 0) atomicMax(slot, __float_as_uint(red[0]));
}

__device__ __forceinline__ signed char quant1(float v, float inv) {
    int q = __float2int_rn(v * inv);
    q = q > 127 ? 127 : (q < -127 ? -127 : q);
    return (signed char)q;
}

__global__ __launch_bounds__(256) void quant_x_kernel(const float4* __restrict__ x,
                                                      char4* __restrict__ q,
                                                      long n4,
                                                      const unsigned* __restrict__ slot) {
    float scale = __uint_as_float(*slot) + EPS;
    float inv = QN / scale;
    long stride = (long)gridDim.x * blockDim.x;
    for (long i = (long)blockIdx.x * blockDim.x + threadIdx.x; i < n4; i += stride) {
        float4 v = x[i];
        char4 c;
        c.x = quant1(v.x, inv);
        c.y = quant1(v.y, inv);
        c.z = quant1(v.z, inv);
        c.w = quant1(v.w, inv);
        q[i] = c;
    }
}

__global__ __launch_bounds__(1024) void quant_w_kernel(const float* __restrict__ W,
                                                       int8_t* __restrict__ Wt,
                                                       const unsigned* __restrict__ slot,
                                                       int K, int F) {
    __shared__ signed char tile[32][33];
    float scale = __uint_as_float(*slot) + EPS;
    float inv = QN / scale;
    int n = blockIdx.x * 32 + threadIdx.x;
    int k = blockIdx.y * 32 + threadIdx.y;
    tile[threadIdx.y][threadIdx.x] = quant1(W[(size_t)k * F + n], inv);
    __syncthreads();
    int n2 = blockIdx.x * 32 + threadIdx.y;
    int k2 = blockIdx.y * 32 + threadIdx.x;
    Wt[(size_t)n2 * K + k2] = tile[threadIdx.x][threadIdx.y];
}

__global__ __launch_bounds__(256) void gemm_i8_kernel(const int8_t* __restrict__ Xq,
                                                      const int8_t* __restrict__ Wt,
                                                      const float* __restrict__ bias,
                                                      const unsigned* __restrict__ slots,
                                                      float* __restrict__ out,
                                                      int M, int K, int F) {
    const int lane = threadIdx.x;
    const int wave = threadIdx.y;
    const int l16  = lane & 15;
    const int half = lane >> 4;

    const int m0 = blockIdx.y * 128 + wave * 16;
    const int n0 = blockIdx.x * 64;

    const float sx = __uint_as_float(slots[0]) + EPS;
    const float sw = __uint_as_float(slots[1]) + EPS;
    const float scale = (sx * sw) / (QN * QN);

    const int8_t* arow = Xq + (size_t)(m0 + l16) * K + half * 16;
    const int8_t* bcol = Wt + (size_t)(n0 + l16) * K + half * 16;
    const size_t bstep = (size_t)16 * K;

    v8i acc0 = {0,0,0,0,0,0,0,0};
    v8i acc1 = {0,0,0,0,0,0,0,0};
    v8i acc2 = {0,0,0,0,0,0,0,0};
    v8i acc3 = {0,0,0,0,0,0,0,0};

    for (int k = 0; k < K; k += 64) {
        v8i a;
        {
            const int2* p = (const int2*)(arow + k);
            int2 g0 = p[0];
            int2 g1 = p[1];
            int2 g2 = p[4];
            int2 g3 = p[5];
            a[0] = g0.x; a[1] = g0.y;
            a[2] = g1.x; a[3] = g1.y;
            a[4] = g2.x; a[5] = g2.y;
            a[6] = g3.x; a[7] = g3.y;
        }
        auto loadB = [&](const int8_t* bp) -> v8i {
            const int4* p = (const int4*)bp;
            int4 h0 = p[0];
            int4 h1 = p[2];
            v8i b;
            b[0] = h0.x; b[1] = h0.y; b[2] = h0.z; b[3] = h0.w;
            b[4] = h1.x; b[5] = h1.y; b[6] = h1.z; b[7] = h1.w;
            return b;
        };
        v8i b0 = loadB(bcol + k);
        v8i b1 = loadB(bcol + bstep + k);
        v8i b2 = loadB(bcol + 2 * bstep + k);
        v8i b3 = loadB(bcol + 3 * bstep + k);

        acc0 = __builtin_amdgcn_wmma_i32_16x16x64_iu8(true, a, true, b0, acc0, false, false);
        acc1 = __builtin_amdgcn_wmma_i32_16x16x64_iu8(true, a, true, b1, acc1, false, false);
        acc2 = __builtin_amdgcn_wmma_i32_16x16x64_iu8(true, a, true, b2, acc2, false, false);
        acc3 = __builtin_amdgcn_wmma_i32_16x16x64_iu8(true, a, true, b3, acc3, false, false);
    }

    #pragma unroll
    for (int r = 0; r < 8; ++r) {
        size_t row = (size_t)(m0 + half * 8 + r);
        float* orow = out + row * F + n0 + l16;
        orow[0]  = (float)acc0[r] * scale + bias[n0 + l16];
        orow[16] = (float)acc1[r] * scale + bias[n0 + 16 + l16];
        orow[32] = (float)acc2[r] * scale + bias[n0 + 32 + l16];
        orow[48] = (float)acc3[r] * scale + bias[n0 + 48 + l16];
    }
}

extern "C" void kernel_launch(void* const* d_in, const int* in_sizes, int n_in,
                              void* d_out, int out_size, void* d_ws, size_t ws_size,
                              hipStream_t stream) {
    const float* X    = (const float*)d_in[0];
    const float* W    = (const float*)d_in[1];
    const float* bias = (const float*)d_in[2];
    float* out = (float*)d_out;

    const int K = 4096;
    const int F = in_sizes[2];
    const int M = in_sizes[0] / K;

    uint8_t* ws = (uint8_t*)d_ws;
    unsigned* slots = (unsigned*)ws;
    int8_t* Xq = (int8_t*)(ws + 256);
    int8_t* Wt = Xq + (size_t)M * K;

    hipMemsetAsync(d_ws, 0, 256, stream);

    long nx4 = (long)M * K / 4;
    long nw4 = (long)K * F / 4;
    maxabs_kernel<<<1024, 256, 0, stream>>>((const float4*)X, nx4, slots + 0);
    maxabs_kernel<<<512,  256, 0, stream>>>((const float4*)W, nw4, slots + 1);

    quant_x_kernel<<<2048, 256, 0, stream>>>((const float4*)X, (char4*)Xq, nx4, slots + 0);

    dim3 tb(32, 32), tg(F / 32, K / 32);
    quant_w_kernel<<<tg, tb, 0, stream>>>(W, Wt, slots + 1, K, F);

    dim3 gb(32, 8), gg(F / 64, M / 128);
    gemm_i8_kernel<<<gg, gb, 0, stream>>>(Xq, Wt, bias, slots, out, M, K, F);
}
